// HyperbolicSupervisedContrastiveLoss_74406013436224
// MI455X (gfx1250) — matrix operand order repaired and checked by reading, not compiled, not run on hardware
//
#include <hip/hip_runtime.h>

typedef __attribute__((ext_vector_type(16))) __bf16 v16bf;
typedef __attribute__((ext_vector_type(8)))  __bf16 v8bf;
typedef __attribute__((ext_vector_type(8)))  float  v8f;
typedef __attribute__((ext_vector_type(4)))  float  f4_t;
typedef __attribute__((ext_vector_type(2)))  float  f2_t;

#define N_TOT   4096
#define DIM     256
#define NSAMP   2048
#define CC      0.01f
#define INVT    (1.0f/0.07f)
#define M_FIX   8.0f                 // fixed softmax shift; sim <= ~7.15 provably
#define LOG2E   1.44269504f
// sim = (0.5*dist_sim + 0.5*ang)/T ; dist_sim = -10*ln(r) = -6.9314718*log2(r)
#define K_LOG   (3.46573590f * INVT)  // 49.5105...  (coefficient of log2(r))
#define K_ANG   (0.5f * INVT)         // 7.142857...

union BFPack  { v16bf v; f4_t q[2]; };
union BF8Pack { v8bf  h; f4_t q;    };

// ---------------------------------------------------------------------------
// Kernel 1: fp32 -> bf16 feature convert + per-row (sq_norm, inv_norm) pairs.
// One wave (32 lanes x 8 floats) per 256-wide row.
// ---------------------------------------------------------------------------
__global__ __launch_bounds__(256) void prep_kernel(
    const float* __restrict__ f, __bf16* __restrict__ fb,
    f2_t* __restrict__ sn) {
  const int wave = threadIdx.x >> 5;
  const int lane = threadIdx.x & 31;
  const int row  = blockIdx.x * 8 + wave;
  const float* src = f + (size_t)row * DIM + lane * 8;
  f4_t a = *(const f4_t*)(src);
  f4_t b = *(const f4_t*)(src + 4);
  BF8Pack u;
  float ss = 0.f;
#pragma unroll
  for (int e = 0; e < 4; ++e) {
    u.h[e]     = (__bf16)a[e]; ss += a[e] * a[e];
    u.h[4 + e] = (__bf16)b[e]; ss += b[e] * b[e];
  }
  *(f4_t*)(fb + (size_t)row * DIM + lane * 8) = u.q;
#pragma unroll
  for (int off = 16; off >= 1; off >>= 1) ss += __shfl_xor(ss, off, 32);
  if (lane == 0) {
    f2_t o;
    o.x = ss;
    o.y = __frsqrt_rn(fmaxf(ss, 1e-24f));
    sn[row] = o;
  }
}

// ---------------------------------------------------------------------------
// Kernel 2: fused Gram (WMMA bf16) + Mobius/hyperbolic sim + fixed-shift
// softmax sums + masked positive sums. Block = 16-row strip; 8 waves split
// the 256 column tiles.
// ---------------------------------------------------------------------------
__global__ __launch_bounds__(256) void supcon_kernel(
    const __bf16* __restrict__ fb, const f2_t* __restrict__ sn,
    const int* __restrict__ labels, float* __restrict__ partials) {
  __shared__ float red[16][8][3];

  const int wave = threadIdx.x >> 5;
  const int lane = threadIdx.x & 31;
  const int l16  = lane & 15;
  const int hi   = lane >> 4;           // half-wave select (K/row split per ISA layout)
  const int rowBase = blockIdx.x * 16;

  // Preload A tile (16 rows x 256 K) in WMMA 16-bit A layout:
  // lanes 0-15 = rows, VGPR elems 0-7 = K {0..7 | 8..15}, elems 8-15 = K{16..23 | 24..31}
  BFPack A[8];
  {
    const __bf16* aptr = fb + (size_t)(rowBase + l16) * DIM + hi * 8;
#pragma unroll
    for (int k = 0; k < 8; ++k) {
      A[k].q[0] = *(const f4_t*)(aptr + k * 32);
      A[k].q[1] = *(const f4_t*)(aptr + k * 32 + 16);
    }
  }

  // Per-lane row constants: C/D layout row M = v + 8*hi, col N = l16.
  float a2[8], ivi[8], s[8], p[8], np[8];
  int labi[8], iv[8];
#pragma unroll
  for (int v = 0; v < 8; ++v) {
    const int i = rowBase + v + hi * 8;
    f2_t sni = sn[i];
    a2[v]  = sni.x;
    ivi[v] = sni.y;
    labi[v] = labels[i & (NSAMP - 1)];
    iv[v] = i;
    s[v] = 0.f; p[v] = 0.f; np[v] = 0.f;
  }

  for (int ct = wave; ct < N_TOT / 16; ct += 8) {
    const int   j    = ct * 16 + l16;
    const f2_t  snj  = sn[j];
    const float b2   = snj.x;
    const float ivj  = snj.y;
    const int   labj = labels[j & (NSAMP - 1)];

    // B tile in 16-bit B layout: lane = N, elems 0-7 = K {0..7 | 8..15}, elems 8-15 = K{16..23 | 24..31}
    const __bf16* bptr = fb + (size_t)j * DIM + hi * 8;
    v8f c = {};
#pragma unroll
    for (int k = 0; k < 8; ++k) {
      BFPack B;
      B.q[0] = *(const f4_t*)(bptr + k * 32);
      B.q[1] = *(const f4_t*)(bptr + k * 32 + 16);
      c = __builtin_amdgcn_wmma_f32_16x16x32_bf16(false, A[k].v, false, B.v,
                                                  (short)0, c, false, false);
    }

    // Element-wise hyperbolic similarity + fixed-shift softmax accumulation.
#pragma unroll
    for (int v = 0; v < 8; ++v) {
      const float g   = c[v];
      const float ab  = -g;
      const float t2  = fmaf(-2.f * CC, g, 1.f);            // 1 + 2c*ab
      const float Am  = fmaf(CC, b2, t2);                   // A
      const float Bm  = fmaf(-CC, a2[v], 1.f);              // B
      const float den = fmaf(CC * CC * a2[v], b2, t2);      // denominator
      float nsq = fmaf(Am * Am, a2[v],
                  fmaf(2.f * Am * Bm, ab, Bm * Bm * b2));   // num_sq
      nsq = fmaxf(nsq * CC, 1e-14f);                        // fold sqrt_c^2 in
      float x = __builtin_amdgcn_sqrtf(nsq) * __builtin_amdgcn_rcpf(den);
      x = fminf(x, 0.99999f);                               // 1 - 1e-5 clamp
      const float lr  = __builtin_amdgcn_logf(1.f + x) -
                        __builtin_amdgcn_logf(1.f - x);     // log2 ratio
      const float ang = g * ivi[v] * ivj;
      const float sim = fmaf(K_ANG, ang, -K_LOG * lr);

      const bool  nd = (j != iv[v]);
      const float e  = __builtin_amdgcn_exp2f(fmaf(sim, LOG2E, -M_FIX * LOG2E));
      s[v] += nd ? e : 0.f;
      const float w = (nd && (labj == labi[v])) ? 1.f : 0.f;
      p[v]  = fmaf(w, sim, p[v]);
      np[v] += w;
    }
  }

  // Plain sum over the 16 column lanes of each half-wave.
#pragma unroll
  for (int off = 1; off < 16; off <<= 1) {
#pragma unroll
    for (int v = 0; v < 8; ++v) {
      s[v]  += __shfl_xor(s[v],  off, 32);
      p[v]  += __shfl_xor(p[v],  off, 32);
      np[v] += __shfl_xor(np[v], off, 32);
    }
  }

  if (l16 == 0) {
#pragma unroll
    for (int v = 0; v < 8; ++v) {
      const int r = v + hi * 8;
      red[r][wave][0] = s[v];
      red[r][wave][1] = p[v];
      red[r][wave][2] = np[v];
    }
  }
  __syncthreads();

  if (threadIdx.x < 16) {
    const int r = threadIdx.x;
    float S = 0.f, P = 0.f, N = 0.f;
#pragma unroll
    for (int w = 0; w < 8; ++w) {
      S += red[r][w][0];
      P += red[r][w][1];
      N += red[r][w][2];
    }
    const float logZ = __logf(S + 1e-10f);
    const float npos = (N < 1e-6f) ? 1.f : N;
    // sum_pos((sim - M) - logZ) = P - N*(M + logZ) ; loss = -(T/BT) * mean
    float loss = -((P - N * (M_FIX + logZ)) / npos);
    loss += __shfl_xor(loss, 1, 32);
    loss += __shfl_xor(loss, 2, 32);
    loss += __shfl_xor(loss, 4, 32);
    loss += __shfl_xor(loss, 8, 32);
    if (r == 0) partials[blockIdx.x] = loss;
  }
}

// ---------------------------------------------------------------------------
// Kernel 3: reduce 256 block partials -> scalar mean.
// ---------------------------------------------------------------------------
__global__ __launch_bounds__(256) void finalize_kernel(
    const float* __restrict__ partials, float* __restrict__ out) {
  __shared__ float acc[8];
  float v = partials[threadIdx.x];
#pragma unroll
  for (int off = 16; off >= 1; off >>= 1) v += __shfl_xor(v, off, 32);
  if ((threadIdx.x & 31) == 0) acc[threadIdx.x >> 5] = v;
  __syncthreads();
  if (threadIdx.x == 0) {
    float t = 0.f;
#pragma unroll
    for (int w = 0; w < 8; ++w) t += acc[w];
    out[0] = t * (1.0f / (float)N_TOT);
  }
}

extern "C" void kernel_launch(void* const* d_in, const int* in_sizes, int n_in,
                              void* d_out, int out_size, void* d_ws, size_t ws_size,
                              hipStream_t stream) {
  const float* features = (const float*)d_in[0];
  const int*   labels   = (const int*)d_in[1];
  float*       out      = (float*)d_out;

  char* ws = (char*)d_ws;
  __bf16* fb       = (__bf16*)ws;                                  // 2 MB bf16 features
  f2_t*   sn       = (f2_t*)(ws + (size_t)N_TOT * DIM * sizeof(__bf16));
  float*  partials = (float*)(sn + N_TOT);

  prep_kernel<<<N_TOT / 8, 256, 0, stream>>>(features, fb, sn);
  supcon_kernel<<<N_TOT / 16, 256, 0, stream>>>(fb, sn, labels, partials);
  finalize_kernel<<<1, 256, 0, stream>>>(partials, out);

  (void)in_sizes; (void)n_in; (void)out_size; (void)ws_size;
}
